// GraphSAGE_40742059770457
// MI455X (gfx1250) — matrix operand order repaired and checked by reading, not compiled, not run on hardware
//
#include <hip/hip_runtime.h>
#include <hip/hip_bf16.h>

typedef __attribute__((ext_vector_type(16))) __bf16 v16bf;
typedef __attribute__((ext_vector_type(8)))  __bf16 v8bf;
typedef __attribute__((ext_vector_type(8)))  float  v8f;
typedef __attribute__((ext_vector_type(4)))  unsigned int u32x4;
typedef __attribute__((ext_vector_type(8)))  int          i32x8;
typedef __attribute__((ext_vector_type(4)))  int          i32x4;

#define IN_F  128
#define HID_F 64
#define OUT_F 64

#if defined(__has_builtin)
#if __has_builtin(__builtin_amdgcn_tensor_load_to_lds) && __has_builtin(__builtin_amdgcn_s_wait_tensorcnt)
#define HAVE_TDM 1
#endif
#endif
#ifndef HAVE_TDM
#define HAVE_TDM 0
#endif

static __device__ __forceinline__ void atomic_add_f32(float* p, float v) {
  __hip_atomic_fetch_add(p, v, __ATOMIC_RELAXED, __HIP_MEMORY_SCOPE_AGENT);
}

#if HAVE_TDM
// 2D TDM load: copy a [rows x width] f32 tile (row stride = stride dwords) from
// global into LDS at byte offset lds_addr. Rows past tensor_rows zero-fill.
// D# bit packing per CDNA5 ISA ch.8; 6-arg builtin form on this toolchain.
static __device__ __forceinline__ void tdm_load_2d(unsigned int lds_addr, const void* gptr,
                                                   unsigned int width, unsigned int tile_rows,
                                                   unsigned int stride, unsigned int tensor_rows) {
  unsigned long long ga = (unsigned long long)(uintptr_t)gptr;
  u32x4 g0;
  g0[0] = 1u;                                                   // count=1 (valid user D#)
  g0[1] = lds_addr;                                             // lds_addr (bytes)
  g0[2] = (unsigned int)ga;                                     // global_addr[31:0]
  g0[3] = (unsigned int)((ga >> 32) & 0x01FFFFFFu) | (2u << 30);// global_addr[56:32] | type=2
  i32x8 g1;
  g1[0] = (int)(2u << 16);                                      // data_size=2 (4 bytes), mask=0
  g1[1] = (int)((width & 0xFFFFu) << 16);                       // tensor_dim0[15:0]
  g1[2] = (int)(((width >> 16) & 0xFFFFu) |
                ((tensor_rows & 0xFFFFu) << 16));               // dim0[31:16] | dim1[15:0]
  g1[3] = (int)(((tensor_rows >> 16) & 0xFFFFu) |
                ((width & 0xFFFFu) << 16));                     // dim1[31:16] | tile_dim0
  g1[4] = (int)(tile_rows & 0xFFFFu);                           // tile_dim1 | tile_dim2=0
  g1[5] = (int)stride;                                          // tensor_dim0_stride[31:0]
  g1[6] = 0;
  g1[7] = 0;
  i32x4 z4 = {0, 0, 0, 0};
  i32x8 z8 = {0, 0, 0, 0, 0, 0, 0, 0};
  __builtin_amdgcn_tensor_load_to_lds(g0, g1, z4, z4, z8, 0);
}
#endif

// ---------------- degree ----------------
__global__ void deg_count(const int* __restrict__ dst, int nE, float* __restrict__ deg) {
  int e = blockIdx.x * blockDim.x + threadIdx.x;
  if (e < nE) atomic_add_f32(&deg[dst[e]], 1.0f);
}

__global__ void deg_inv(float* __restrict__ deg, int n) {
  int i = blockIdx.x * blockDim.x + threadIdx.x;
  if (i < n) deg[i] = 1.0f / fmaxf(deg[i], 1.0f);
}

// ---------------- edge scatter-add (float4 per thread) ----------------
template <int F>
__global__ void scatter_add(const float* __restrict__ h, const int* __restrict__ src,
                            const int* __restrict__ dst, float* __restrict__ neigh, int nE) {
  const int CH = F / 4;
  long long gid = (long long)blockIdx.x * blockDim.x + threadIdx.x;
  int e = (int)(gid / CH);
  if (e >= nE) return;
  int c = ((int)(gid % CH)) * 4;
  const float4 v = *(const float4*)(h + (size_t)src[e] * F + c);
  float* p = neigh + (size_t)dst[e] * F + c;
  atomic_add_f32(p + 0, v.x);
  atomic_add_f32(p + 1, v.y);
  atomic_add_f32(p + 2, v.z);
  atomic_add_f32(p + 3, v.w);
}

// ---------------- weight prep: bf16, column-major, K-concatenated ----------------
__global__ void prep_weights(const float* __restrict__ Ws, const float* __restrict__ Wn,
                             unsigned short* __restrict__ Wt, int KH) {
  int KT = 2 * KH;
  int idx = blockIdx.x * blockDim.x + threadIdx.x;
  if (idx >= 64 * KT) return;
  int n = idx / KT, k = idx % KT;
  float v = (k < KH) ? Ws[(size_t)k * 64 + n] : Wn[(size_t)(k - KH) * 64 + n];
  __bf16 b = (__bf16)v;
  Wt[idx] = __builtin_bit_cast(unsigned short, b);
}

// ---------------- fused SAGE GEMM: out = relu([A_self | A_neigh*invdeg] @ Wt^T + b)
// KBLK = 32-wide K blocks per half. Block = 128 thr = 4 waves, tile = 16 rows x 64 cols.
// Stage: TDM drops raw f32 panels -> one cooperative scale+cvt pass to bf16 in LDS
// -> hot loop is pure {2x ds_load_b128 (A), 2x global_load_b128 (B), 1x v_wmma}.
template <int KBLK>
__global__ __launch_bounds__(128)
void sage_gemm(const float* __restrict__ Aself, const float* __restrict__ Aneigh,
               const float* __restrict__ invdeg, const unsigned short* __restrict__ Wt_,
               const float* __restrict__ bias, float* __restrict__ out, int nN) {
  constexpr int LDA = KBLK * 32;    // K per half
  constexpr int KTOT = 2 * LDA;
  constexpr int PANEL = 16 * LDA;   // floats per panel
  __shared__ __align__(16) float smemRaw[2 * PANEL];
  __shared__ __align__(16) unsigned short smemB_[2 * PANEL];
  __bf16* smemB = reinterpret_cast<__bf16*>(smemB_);

  const __bf16* __restrict__ Wt = reinterpret_cast<const __bf16*>(Wt_);
  const int lane = threadIdx.x & 31;
  const int wave = threadIdx.x >> 5;
  const int half16 = lane >> 4;
  const int rowBase = blockIdx.x * 16;
  const int col = wave * 16 + (lane & 15);

  // ---- 1) raw f32 panels (self + neigh) into LDS
#if HAVE_TDM
  if (wave == 0) {
    unsigned int lds0 = (unsigned int)(uintptr_t)&smemRaw[0];
    tdm_load_2d(lds0, Aself + (size_t)rowBase * LDA, LDA, 16, LDA,
                (unsigned int)(nN - rowBase));
    tdm_load_2d(lds0 + (unsigned int)PANEL * 4u, Aneigh + (size_t)rowBase * LDA, LDA, 16, LDA,
                (unsigned int)(nN - rowBase));
    __builtin_amdgcn_s_wait_tensorcnt(0);
  }
  __syncthreads();
#else
  for (int i = threadIdx.x; i < 2 * PANEL / 4; i += 128) {
    int panel = i / (PANEL / 4);
    int rem = i - panel * (PANEL / 4);
    int r = (rem * 4) / LDA, c = (rem * 4) % LDA;
    int row = min(rowBase + r, nN - 1);
    const float* srcp = (panel ? Aneigh : Aself) + (size_t)row * LDA + c;
    *(float4*)&smemRaw[i * 4] = *(const float4*)srcp;
  }
  __syncthreads();
#endif

  // ---- 2) one cooperative scale + f32->bf16 conversion pass (done ONCE per block)
  {
    constexpr int VPT = 2 * PANEL / 128;        // floats per thread (32 or 16), contiguous
    const int base = threadIdx.x * VPT;         // stays within one row (VPT <= LDA)
    const int r = base / LDA;                   // 0..31: panel*16 + rowInPanel
    const float s = (r >= 16) ? invdeg[min(rowBase + (r & 15), nN - 1)] : 1.0f;
#pragma unroll
    for (int j = 0; j < VPT; j += 8) {
      float4 f0 = *(const float4*)&smemRaw[base + j];
      float4 f1 = *(const float4*)&smemRaw[base + j + 4];
      v8bf o;
      o[0] = (__bf16)(f0.x * s); o[1] = (__bf16)(f0.y * s);
      o[2] = (__bf16)(f0.z * s); o[3] = (__bf16)(f0.w * s);
      o[4] = (__bf16)(f1.x * s); o[5] = (__bf16)(f1.y * s);
      o[6] = (__bf16)(f1.z * s); o[7] = (__bf16)(f1.w * s);
      *(v8bf*)&smemB[base + j] = o;
    }
  }
  __syncthreads();

  // ---- 3) hot loop: pure data movement + WMMA
  v8f acc = {};
#pragma unroll
  for (int half = 0; half < 2; ++half) {
#pragma unroll
    for (int kb = 0; kb < KBLK; ++kb) {
      const int k0 = kb * 32;
      // A fragment: two 8-bf16 runs (one b128 each) from converted LDS panel
      const __bf16* arow = smemB + half * PANEL + (lane & 15) * LDA + k0 + half16 * 8;
      v8bf alo = *(const v8bf*)(arow + 0);
      v8bf ahi = *(const v8bf*)(arow + 16);
      v16bf av = __builtin_shufflevector(alo, ahi, 0, 1, 2, 3, 4, 5, 6, 7,
                                         8, 9, 10, 11, 12, 13, 14, 15);
      // B fragment: two 8-bf16 runs (one b128 each) from transposed weights
      const __bf16* wc = Wt + (size_t)col * KTOT + half * LDA + k0 + half16 * 8;
      v8bf blo = *(const v8bf*)(wc + 0);
      v8bf bhi = *(const v8bf*)(wc + 16);
      v16bf bv = __builtin_shufflevector(blo, bhi, 0, 1, 2, 3, 4, 5, 6, 7,
                                         8, 9, 10, 11, 12, 13, 14, 15);
      acc = __builtin_amdgcn_wmma_f32_16x16x32_bf16(
          false, av, false, bv, (short)0, acc, false, false);
    }
  }

  // ---- epilogue: bias (per column), ReLU, store
  const float bc = bias[col];
  const int mbase = half16 * 8;
#pragma unroll
  for (int r = 0; r < 8; ++r) {
    int orow = rowBase + mbase + r;
    float v = acc[r] + bc;
    v = v > 0.0f ? v : 0.0f;
    if (orow < nN) out[(size_t)orow * HID_F + col] = v;
  }
}

extern "C" void kernel_launch(void* const* d_in, const int* in_sizes, int n_in,
                              void* d_out, int out_size, void* d_ws, size_t ws_size,
                              hipStream_t stream) {
  const float* x   = (const float*)d_in[0];
  const int*   src = (const int*)d_in[1];
  const int*   dst = (const int*)d_in[2];
  const float* W1s = (const float*)d_in[3];
  const float* W1n = (const float*)d_in[4];
  const float* b1  = (const float*)d_in[5];
  const float* W2s = (const float*)d_in[6];
  const float* W2n = (const float*)d_in[7];
  const float* b2  = (const float*)d_in[8];
  float* out = (float*)d_out;

  const int nE = in_sizes[1];
  const int nN = in_sizes[0] / IN_F;

  // ws layout: [deg nN][neigh1 nN*128][neigh2 nN*64][h1 nN*64][Wt1 64*256 bf16][Wt2 64*128 bf16]
  float* deg    = (float*)d_ws;
  float* neigh1 = deg + nN;
  float* neigh2 = neigh1 + (size_t)nN * IN_F;
  float* h1     = neigh2 + (size_t)nN * HID_F;
  unsigned short* Wt1 = (unsigned short*)(h1 + (size_t)nN * HID_F);
  unsigned short* Wt2 = Wt1 + 64 * 2 * IN_F;

  // zero deg + neigh1 + neigh2 (contiguous); h1/Wt fully overwritten
  size_t zbytes = (size_t)nN * (1 + IN_F + HID_F) * sizeof(float);
  (void)hipMemsetAsync(d_ws, 0, zbytes, stream);

  prep_weights<<<(64 * 2 * IN_F + 255) / 256, 256, 0, stream>>>(W1s, W1n, Wt1, IN_F);
  prep_weights<<<(64 * 2 * HID_F + 255) / 256, 256, 0, stream>>>(W2s, W2n, Wt2, HID_F);

  deg_count<<<(nE + 255) / 256, 256, 0, stream>>>(dst, nE, deg);
  deg_inv<<<(nN + 255) / 256, 256, 0, stream>>>(deg, nN);

  {
    long long t = (long long)nE * (IN_F / 4);
    scatter_add<IN_F><<<(unsigned)((t + 255) / 256), 256, 0, stream>>>(x, src, dst, neigh1, nE);
  }
  sage_gemm<IN_F / 32><<<(nN + 15) / 16, 128, 0, stream>>>(x, neigh1, deg, Wt1, b1, h1, nN);
  {
    long long t = (long long)nE * (HID_F / 4);
    scatter_add<HID_F><<<(unsigned)((t + 255) / 256), 256, 0, stream>>>(h1, src, dst, neigh2, nE);
  }
  sage_gemm<HID_F / 32><<<(nN + 15) / 16, 128, 0, stream>>>(h1, neigh2, deg, Wt2, b2, out, nN);
}
